// GraphHead_14293651161366
// MI455X (gfx1250) — matrix operand order repaired and checked by reading, not compiled, not run on hardware
//
#include <hip/hip_runtime.h>
#include <hip/hip_bf16.h>

// ---------------------------------------------------------------------------
// CDNA5 / gfx1250 implementation. wave32, WMMA 16x16x32 f16 -> f32.
// ---------------------------------------------------------------------------

typedef __attribute__((ext_vector_type(16))) _Float16 v16h;
typedef __attribute__((ext_vector_type(2)))  __fp16   v2fp16;  // cvt_pkrtz result type
typedef __attribute__((ext_vector_type(8)))  float    v8f;

#define NHEADS 48
#define NBOX   384
#define P_TOT  (NHEADS * NBOX)   // 18432
#define CARDN  16
#define SUBD   64
#define REPRD  1024
#define ENCD   1024

#define WMMA_F16(a, b, c) \
  __builtin_amdgcn_wmma_f32_16x16x32_f16(false, (a), false, (b), (short)0, (c), false, false)

// packed f32x2 -> f16x2 convert + 32-bit LDS store
__device__ __forceinline__ void store_pk(_Float16* dst, float lo, float hi) {
  union { v2fp16 h; unsigned u; } cv;
  cv.h = __builtin_amdgcn_cvt_pkrtz(lo, hi);
  *(unsigned*)dst = cv.u;
}

__device__ __forceinline__ unsigned pk_u32(float lo, float hi) {
  union { v2fp16 h; unsigned u; } cv;
  cv.h = __builtin_amdgcn_cvt_pkrtz(lo, hi);
  return cv.u;
}

// 16-bit A-matrix 16x32 fragment. K values come in contiguous
// pairs per VGPR: pair p -> K = {16h+2p, 16h+2p+1} (matches the B loaders),
// so the fragment is 8 aligned 32-bit LDS reads from a row-major tile.
__device__ __forceinline__ v16h load_a_frag(const _Float16* s, int stride) {
  int lane = threadIdx.x & 31;
  int m = lane & 15, half = lane >> 4;
  union { v16h v; unsigned u[8]; } fa;
#pragma unroll
  for (int p = 0; p < 8; ++p) {
    int bk = 16 * half + 2 * p;
    fa.u[p] = *(const unsigned*)(s + m * stride + bk);
  }
  return fa.v;
}

// B 32x16 fragment from a COLUMN-MAJOR LDS tile (s[col*stride + k]):
// lane n = lane&15 reads K = 16*half + e, e contiguous -> 8 u32 reads.
__device__ __forceinline__ v16h load_b_frag_t(const _Float16* s, int stride, int col0) {
  int lane = threadIdx.x & 31;
  int n = lane & 15, half = lane >> 4;
  union { v16h v; unsigned u[8]; } fb;
  const _Float16* base = s + (col0 + n) * stride + 16 * half;
#pragma unroll
  for (int p = 0; p < 8; ++p) fb.u[p] = *(const unsigned*)(base + 2 * p);
  return fb.v;
}

// B 32x16 fragment straight from global f32 (row-major, stride = N):
// row pairs converted with one packed cvt each.
__device__ __forceinline__ v16h load_b_frag_g(const float* g, int stride, int row0, int col0) {
  int lane = threadIdx.x & 31;
  int n = lane & 15, half = lane >> 4;
  union { v16h v; unsigned u[8]; } fb;
  const float* base = g + (long)(row0 + 16 * half) * stride + col0 + n;
#pragma unroll
  for (int p = 0; p < 8; ++p) {
    float lo = base[(long)(2 * p) * stride];
    float hi = base[(long)(2 * p + 1) * stride];
    fb.u[p] = pk_u32(lo, hi);
  }
  return fb.v;
}

// ---------------------------------------------------------------------------
// Tiled WMMA GEMM:  C = [relu](A(MxK) @ B(KxN) + bias)
// block = 256 threads (8 waves), tile 32x128, K-step 32.
// Requires M%32==0 and N%128==0 (true for all launches); K tail handled.
// ---------------------------------------------------------------------------
#define GM 32
#define GN 128
#define GK 32
#define LDA 40    // halfs per A row
#define LDBK 40   // halfs per B column (col-major LDS)

__global__ __launch_bounds__(256) void gemm_bias_relu(
    const float* __restrict__ Am, const float* __restrict__ Bm,
    const float* __restrict__ bias, float* __restrict__ Cm,
    int M, int N, int K, int do_relu)
{
  __shared__ _Float16 sA[GM * LDA];     // row-major 32 x 32
  __shared__ _Float16 sBt[GN * LDBK];   // col-major 128 cols x 32 k
  int tid = threadIdx.x;
  int wave = tid >> 5;
  int mi = wave >> 2;   // 0..1
  int ni = wave & 3;    // 0..3
  int m0 = blockIdx.y * GM;
  int n0 = blockIdx.x * GN;

  v8f acc0 = {}; v8f acc1 = {};
  int kfull = K & ~(GK - 1);

  for (int k0 = 0; k0 < kfull; k0 += GK) {
#pragma unroll
    for (int q = 0; q < 2; ++q) {           // A: 512 k-pairs
      int i = tid + q * 256;
      int r = i >> 4, pj = i & 15;
      const float* src = Am + (long)(m0 + r) * K + k0 + 2 * pj;
      store_pk(&sA[r * LDA + 2 * pj], src[0], src[1]);
    }
#pragma unroll
    for (int q = 0; q < 8; ++q) {           // B: 2048 (col, k-pair)
      int i = tid + q * 256;
      int kp = i >> 7, col = i & 127;
      const float* src = Bm + (long)(k0 + 2 * kp) * N + n0 + col;
      store_pk(&sBt[col * LDBK + 2 * kp], src[0], src[N]);
    }
    __syncthreads();
    v16h a  = load_a_frag(&sA[mi * 16 * LDA], LDA);
    v16h b0 = load_b_frag_t(sBt, LDBK, ni * 32);
    v16h b1 = load_b_frag_t(sBt, LDBK, ni * 32 + 16);
    acc0 = WMMA_F16(a, b0, acc0);
    acc1 = WMMA_F16(a, b1, acc1);
    __syncthreads();
  }
  if (kfull < K) {                          // zero-padded tail (K = 36 case)
#pragma unroll
    for (int q = 0; q < 2; ++q) {
      int i = tid + q * 256;
      int r = i >> 4, pj = i & 15;
      int kc = kfull + 2 * pj;
      const float* src = Am + (long)(m0 + r) * K;
      float lo = (kc < K) ? src[kc] : 0.f;
      float hi = (kc + 1 < K) ? src[kc + 1] : 0.f;
      store_pk(&sA[r * LDA + 2 * pj], lo, hi);
    }
#pragma unroll
    for (int q = 0; q < 8; ++q) {
      int i = tid + q * 256;
      int kp = i >> 7, col = i & 127;
      int kc = kfull + 2 * kp;
      float lo = (kc < K) ? Bm[(long)kc * N + n0 + col] : 0.f;
      float hi = (kc + 1 < K) ? Bm[(long)(kc + 1) * N + n0 + col] : 0.f;
      store_pk(&sBt[col * LDBK + 2 * kp], lo, hi);
    }
    __syncthreads();
    v16h a  = load_a_frag(&sA[mi * 16 * LDA], LDA);
    v16h b0 = load_b_frag_t(sBt, LDBK, ni * 32);
    v16h b1 = load_b_frag_t(sBt, LDBK, ni * 32 + 16);
    acc0 = WMMA_F16(a, b0, acc0);
    acc1 = WMMA_F16(a, b1, acc1);
    __syncthreads();
  }

  int lane = tid & 31;
  int n = lane & 15, half = lane >> 4;
  int colA = n0 + ni * 32 + n;
  int colB = colA + 16;
  float bA = bias[colA], bB = bias[colB];
#pragma unroll
  for (int r = 0; r < 8; ++r) {
    int row = m0 + mi * 16 + r + 8 * half;
    float vA = acc0[r] + bA;
    float vB = acc1[r] + bB;
    if (do_relu) { vA = fmaxf(vA, 0.f); vB = fmaxf(vB, 0.f); }
    Cm[(long)row * N + colA] = vA;
    Cm[(long)row * N + colB] = vB;
  }
}

// ---------------------------------------------------------------------------
// Fused per-card attention/message kernel.
// One block = 32 P-rows x 1024 out cols, 8 waves.
//   per card: waves 0-3 compute u (app @ W1[c] + b1) for both 16-row M-tiles,
//   waves 4-7 compute v (sp @ W2[c] + b2); t = relu(u*v) -> LDS f16;
//   all waves accumulate t @ W3[c] into resident f32 tiles (2 Mtiles x 8
//   Ntiles each). Epilogue adds b3sum, optional relu, optional diagonal
//   row compaction into the concatenated output.
// mode: 0 = pair concat (appA[x] | appB[y]), 1 = gather y, 2 = gather x,
//       3 = broadcast single row (gfeat)
// ---------------------------------------------------------------------------
__global__ __launch_bounds__(256) void attn_fused(
    int mode,
    const float* __restrict__ appA, const float* __restrict__ appB, int Ea,
    const float* __restrict__ sp,
    const float* __restrict__ W1, const float* __restrict__ b1,
    const float* __restrict__ W2, const float* __restrict__ b2,
    const float* __restrict__ W3, const float* __restrict__ b3sum,
    float* __restrict__ out, int out_stride, int out_col0,
    int do_relu, int compact)
{
  __shared__ _Float16 sApp[32 * 40];    // row-major 32 x 32
  __shared__ _Float16 sSp[32 * 40];
  __shared__ _Float16 sW1t[64 * 40];    // col-major 64 cols x 32 k
  __shared__ _Float16 sW2t[64 * 40];
  __shared__ float    uL[32 * 64];
  __shared__ float    vL[32 * 64];
  __shared__ _Float16 tL[32 * 72];      // row-major 32 x 64 (+pad)

  int tid = threadIdx.x;
  int wave = tid >> 5;
  int lane = tid & 31;
  int p0 = blockIdx.x * 32;

  v8f acc[2][8];
#pragma unroll
  for (int mt = 0; mt < 2; ++mt)
#pragma unroll
    for (int i = 0; i < 8; ++i) acc[mt][i] = (v8f){};

  int Kmax = (Ea > ENCD) ? Ea : ENCD;

  for (int c = 0; c < CARDN; ++c) {
    v8f uv0 = {}; v8f uv1 = {};
    for (int k0 = 0; k0 < Kmax; k0 += 32) {
      if (k0 < Ea) {
        // app tile: 512 k-pairs, 2 per thread
#pragma unroll
        for (int q = 0; q < 2; ++q) {
          int i = tid + q * 256;
          int r = i >> 4, pj = i & 15;
          int p = p0 + r;
          const float* rowp;
          if (mode == 0) {
            int x = p / NBOX, y = p - x * NBOX;
            rowp = (k0 < ENCD) ? appA + (long)x * ENCD + k0
                               : appB + (long)y * ENCD + (k0 - ENCD);
          } else if (mode == 1) {
            rowp = appA + (long)(p % NBOX) * ENCD + k0;
          } else if (mode == 2) {
            rowp = appA + (long)(p / NBOX) * ENCD + k0;
          } else {
            rowp = appA + k0;
          }
          store_pk(&sApp[r * 40 + 2 * pj], rowp[2 * pj], rowp[2 * pj + 1]);
        }
        // W1 tile 32x64 -> col-major: 1024 (col, k-pair), 4 per thread
#pragma unroll
        for (int q = 0; q < 4; ++q) {
          int i = tid + q * 256;
          int kp = i >> 6, col = i & 63;
          const float* src = W1 + ((long)c * Ea + k0 + 2 * kp) * SUBD + col;
          store_pk(&sW1t[col * 40 + 2 * kp], src[0], src[SUBD]);
        }
      }
      if (k0 < ENCD) {
#pragma unroll
        for (int q = 0; q < 2; ++q) {
          int i = tid + q * 256;
          int r = i >> 4, pj = i & 15;
          const float* rowp = sp + (long)(p0 + r) * ENCD + k0;
          store_pk(&sSp[r * 40 + 2 * pj], rowp[2 * pj], rowp[2 * pj + 1]);
        }
#pragma unroll
        for (int q = 0; q < 4; ++q) {
          int i = tid + q * 256;
          int kp = i >> 6, col = i & 63;
          const float* src = W2 + ((long)c * ENCD + k0 + 2 * kp) * SUBD + col;
          store_pk(&sW2t[col * 40 + 2 * kp], src[0], src[SUBD]);
        }
      }
      __syncthreads();
      if (wave < 4) {
        if (k0 < Ea) {
          v16h b  = load_b_frag_t(sW1t, 40, wave * 16);
          v16h a0 = load_a_frag(sApp, 40);
          v16h a1 = load_a_frag(&sApp[16 * 40], 40);
          uv0 = WMMA_F16(a0, b, uv0);
          uv1 = WMMA_F16(a1, b, uv1);
        }
      } else {
        if (k0 < ENCD) {
          v16h b  = load_b_frag_t(sW2t, 40, (wave - 4) * 16);
          v16h a0 = load_a_frag(sSp, 40);
          v16h a1 = load_a_frag(&sSp[16 * 40], 40);
          uv0 = WMMA_F16(a0, b, uv0);
          uv1 = WMMA_F16(a1, b, uv1);
        }
      }
      __syncthreads();
    }
    // scatter u/v (+bias) to LDS, row-major f32
    {
      int n = lane & 15, half = lane >> 4;
      int col = (wave & 3) * 16 + n;
      float bb = ((wave < 4) ? b1 : b2)[c * SUBD + col];
      float* dst = (wave < 4) ? uL : vL;
#pragma unroll
      for (int r = 0; r < 8; ++r) {
        dst[(r + 8 * half) * 64 + col] = uv0[r] + bb;
        dst[(16 + r + 8 * half) * 64 + col] = uv1[r] + bb;
      }
    }
    __syncthreads();
    // t = relu(u*v), packed f16
#pragma unroll
    for (int q = 0; q < 4; ++q) {
      int i = tid + q * 256;          // 1024 pairs
      int r = i >> 5, s2 = i & 31;
      float a0 = uL[r * 64 + 2 * s2] * vL[r * 64 + 2 * s2];
      float a1 = uL[r * 64 + 2 * s2 + 1] * vL[r * 64 + 2 * s2 + 1];
      store_pk(&tL[r * 72 + 2 * s2], fmaxf(a0, 0.f), fmaxf(a1, 0.f));
    }
    __syncthreads();
    // acc += t(32x64) @ W3[c](64 x 128 cols per wave)
    {
      const float* W3c = W3 + (long)c * SUBD * REPRD;
      v16h a00 = load_a_frag(tL, 72);
      v16h a01 = load_a_frag(tL + 32, 72);
      v16h a10 = load_a_frag(&tL[16 * 72], 72);
      v16h a11 = load_a_frag(&tL[16 * 72] + 32, 72);
#pragma unroll
      for (int nt = 0; nt < 8; ++nt) {
        int col0 = wave * 128 + nt * 16;
        v16h b0 = load_b_frag_g(W3c, REPRD, 0,  col0);
        v16h b1 = load_b_frag_g(W3c, REPRD, 32, col0);
        acc[0][nt] = WMMA_F16(a00, b0, acc[0][nt]);
        acc[0][nt] = WMMA_F16(a01, b1, acc[0][nt]);
        acc[1][nt] = WMMA_F16(a10, b0, acc[1][nt]);
        acc[1][nt] = WMMA_F16(a11, b1, acc[1][nt]);
      }
    }
    __syncthreads();
  }

  // epilogue
  int n = lane & 15, half = lane >> 4;
#pragma unroll
  for (int mt = 0; mt < 2; ++mt) {
#pragma unroll
    for (int r = 0; r < 8; ++r) {
      int p = p0 + mt * 16 + r + 8 * half;
      int orow = p;
      bool write = true;
      if (compact) {
        int x = p / NBOX, y = p - x * NBOX;
        if (x == y) write = false;
        int q = (p + NBOX) / (NBOX + 1);   // ceil(p/385) diagonals skipped
        if (q > NHEADS) q = NHEADS;
        orow = p - q;
      }
      if (write) {
#pragma unroll
        for (int nt = 0; nt < 8; ++nt) {
          int col = wave * 128 + nt * 16 + n;
          float v = acc[mt][nt][r] + b3sum[col];
          if (do_relu) v = fmaxf(v, 0.f);
          out[(long)orow * out_stride + out_col0 + col] = v;
        }
      }
    }
  }
}

// ---------------------------------------------------------------------------
// Small helper kernels
// ---------------------------------------------------------------------------
__global__ void gfeat_kernel(const float* __restrict__ feat3, float* __restrict__ gfeat) {
  int c = threadIdx.x;   // 256 channels
  float s = 0.f;
  for (int i = 0; i < 625; ++i) s += feat3[c * 625 + i];
  gfeat[c] = s * (1.f / 625.f);
}

__global__ void colsum16_kernel(const float* __restrict__ b, float* __restrict__ out, int R) {
  int i = blockIdx.x * 256 + threadIdx.x;
  if (i < R) {
    float s = 0.f;
    for (int c = 0; c < CARDN; ++c) s += b[c * R + i];
    out[i] = s;
  }
}

__global__ __launch_bounds__(256) void adj_kernel(
    const float* __restrict__ w, const float* __restrict__ Wadj,
    const float* __restrict__ badj, float* __restrict__ adj)
{
  int p = blockIdx.x;
  float s = 0.f;
  for (int i = threadIdx.x; i < REPRD; i += 256) s += w[(long)p * REPRD + i] * Wadj[i];
  __shared__ float red[256];
  red[threadIdx.x] = s;
  __syncthreads();
  for (int st = 128; st > 0; st >>= 1) {
    if (threadIdx.x < st) red[threadIdx.x] += red[threadIdx.x + st];
    __syncthreads();
  }
  if (threadIdx.x == 0) adj[p] = red[0] + badj[0];
}

__global__ __launch_bounds__(256) void softmax_rows(
    const float* __restrict__ in, float* __restrict__ out, int C, int rs, int cs)
{
  int r = blockIdx.x;
  __shared__ float red[256];
  float m = -1e30f;
  for (int i = threadIdx.x; i < C; i += 256) m = fmaxf(m, in[(long)r * rs + (long)i * cs]);
  red[threadIdx.x] = m; __syncthreads();
  for (int st = 128; st > 0; st >>= 1) {
    if (threadIdx.x < st) red[threadIdx.x] = fmaxf(red[threadIdx.x], red[threadIdx.x + st]);
    __syncthreads();
  }
  m = red[0]; __syncthreads();
  float s = 0.f;
  for (int i = threadIdx.x; i < C; i += 256) s += expf(in[(long)r * rs + (long)i * cs] - m);
  red[threadIdx.x] = s; __syncthreads();
  for (int st = 128; st > 0; st >>= 1) {
    if (threadIdx.x < st) red[threadIdx.x] += red[threadIdx.x + st];
    __syncthreads();
  }
  s = red[0];
  float invs = 1.f / s;
  for (int i = threadIdx.x; i < C; i += 256)
    out[(long)r * C + i] = expf(in[(long)r * rs + (long)i * cs] - m) * invs;
}

// softmax-weighted message reduction + residual + LayerNorm
// axis==0: blocks over h, msg row = h*384 + j   (m_h)
// axis==1: blocks over n, msg row = j*384 + n   (m_o)
__global__ __launch_bounds__(256) void msg_reduce_ln(
    const float* __restrict__ soft, const float* __restrict__ msg,
    const float* __restrict__ base, const float* __restrict__ g,
    const float* __restrict__ b, float* __restrict__ out, int Cn, int axis)
{
  int rb = blockIdx.x;
  float accv[4] = {0.f, 0.f, 0.f, 0.f};
  for (int j = 0; j < Cn; ++j) {
    float wgt = soft[rb * Cn + j];
    long mrow = (axis == 0) ? ((long)rb * NBOX + j) : ((long)j * NBOX + rb);
    const float* row = msg + mrow * REPRD;
#pragma unroll
    for (int q = 0; q < 4; ++q) accv[q] += wgt * row[threadIdx.x + q * 256];
  }
  __shared__ float srow[REPRD];
#pragma unroll
  for (int q = 0; q < 4; ++q) {
    int col = threadIdx.x + q * 256;
    srow[col] = base[(long)rb * REPRD + col] + fmaxf(accv[q], 0.f);
  }
  __syncthreads();
  float ls = 0.f, ls2 = 0.f;
  for (int col = threadIdx.x; col < REPRD; col += 256) {
    float v = srow[col]; ls += v; ls2 += v * v;
  }
  __shared__ float r1[256], r2[256];
  r1[threadIdx.x] = ls; r2[threadIdx.x] = ls2;
  __syncthreads();
  for (int st = 128; st > 0; st >>= 1) {
    if (threadIdx.x < st) {
      r1[threadIdx.x] += r1[threadIdx.x + st];
      r2[threadIdx.x] += r2[threadIdx.x + st];
    }
    __syncthreads();
  }
  float mu = r1[0] / REPRD;
  float var = r2[0] / REPRD - mu * mu;
  float inv = rsqrtf(var + 1e-5f);
  for (int col = threadIdx.x; col < REPRD; col += 256)
    out[(long)rb * REPRD + col] = (srow[col] - mu) * inv * g[col] + b[col];
}

// ---------------------------------------------------------------------------
// Host launcher
// ---------------------------------------------------------------------------
extern "C" void kernel_launch(void* const* d_in, const int* in_sizes, int n_in,
                              void* d_out, int out_size, void* d_ws, size_t ws_size,
                              hipStream_t stream) {
  (void)in_sizes; (void)n_in; (void)out_size; (void)ws_size;
  const float* box   = (const float*)d_in[0];
  const float* feat3 = (const float*)d_in[1];
  const float* sp36  = (const float*)d_in[2];
  const float* Wbh1  = (const float*)d_in[3];
  const float* bbh1  = (const float*)d_in[4];
  const float* Wbh2  = (const float*)d_in[5];
  const float* bbh2  = (const float*)d_in[6];
  const float* Wsp1  = (const float*)d_in[7];
  const float* bsp1  = (const float*)d_in[8];
  const float* Wsp2  = (const float*)d_in[9];
  const float* bsp2  = (const float*)d_in[10];
  const float* Wsp3  = (const float*)d_in[11];
  const float* bsp3  = (const float*)d_in[12];
  const float* Wadj  = (const float*)d_in[13];
  const float* badj  = (const float*)d_in[14];
  const float* A1W = (const float*)d_in[15]; const float* A1b = (const float*)d_in[16];
  const float* A2W = (const float*)d_in[17]; const float* A2b = (const float*)d_in[18];
  const float* A3W = (const float*)d_in[19]; const float* A3b = (const float*)d_in[20];
  const float* O1W = (const float*)d_in[21]; const float* O1b = (const float*)d_in[22];
  const float* O2W = (const float*)d_in[23]; const float* O2b = (const float*)d_in[24];
  const float* O3W = (const float*)d_in[25]; const float* O3b = (const float*)d_in[26];
  const float* S1W = (const float*)d_in[27]; const float* S1b = (const float*)d_in[28];
  const float* S2W = (const float*)d_in[29]; const float* S2b = (const float*)d_in[30];
  const float* S3W = (const float*)d_in[31]; const float* S3b = (const float*)d_in[32];
  const float* G1W = (const float*)d_in[33]; const float* G1b = (const float*)d_in[34];
  const float* G2W = (const float*)d_in[35]; const float* G2b = (const float*)d_in[36];
  const float* G3W = (const float*)d_in[37]; const float* G3b = (const float*)d_in[38];
  const float* gln_h = (const float*)d_in[39];
  const float* bln_h = (const float*)d_in[40];
  const float* gln_o = (const float*)d_in[41];
  const float* bln_o = (const float*)d_in[42];

  float* out = (float*)d_out;
  float* ws = (float*)d_ws;
  size_t off = 0;
  auto alloc = [&](size_t nf) { float* p = ws + off; off += nf; return p; };

  float* gfeat  = alloc(256);
  float* b3A    = alloc(REPRD);
  float* b3O    = alloc(REPRD);
  float* b3S    = alloc(REPRD);
  float* b3G    = alloc(REPRD);
  float* node1  = alloc((size_t)NBOX * ENCD);
  float* node   = alloc((size_t)NBOX * ENCD);
  float* hnodeL = alloc((size_t)NHEADS * ENCD);
  float* nodeL  = alloc((size_t)NBOX * ENCD);
  float* sp1    = alloc((size_t)P_TOT * 128);
  float* sp2    = alloc((size_t)P_TOT * 256);
  float* sp3    = alloc((size_t)P_TOT * REPRD);
  float* wbuf   = alloc((size_t)P_TOT * REPRD);  // attn pass1 out; reused as msgO
  float* msgS   = alloc((size_t)P_TOT * REPRD);
  float* adj    = alloc(P_TOT);
  float* softA  = alloc((size_t)NHEADS * NBOX);
  float* softB  = alloc((size_t)NBOX * NHEADS);

  gfeat_kernel<<<1, 256, 0, stream>>>(feat3, gfeat);
  colsum16_kernel<<<4, 256, 0, stream>>>(A3b, b3A, REPRD);
  colsum16_kernel<<<4, 256, 0, stream>>>(O3b, b3O, REPRD);
  colsum16_kernel<<<4, 256, 0, stream>>>(S3b, b3S, REPRD);
  colsum16_kernel<<<4, 256, 0, stream>>>(G3b, b3G, REPRD);

  // node MLP
  gemm_bias_relu<<<dim3(ENCD / 128, NBOX / 32), 256, 0, stream>>>(
      box, Wbh1, bbh1, node1, NBOX, ENCD, 12544, 1);
  gemm_bias_relu<<<dim3(ENCD / 128, NBOX / 32), 256, 0, stream>>>(
      node1, Wbh2, bbh2, node, NBOX, ENCD, ENCD, 1);
  // spatial MLP
  gemm_bias_relu<<<dim3(1, P_TOT / 32), 256, 0, stream>>>(sp36, Wsp1, bsp1, sp1, P_TOT, 128, 36, 1);
  gemm_bias_relu<<<dim3(2, P_TOT / 32), 256, 0, stream>>>(sp1, Wsp2, bsp2, sp2, P_TOT, 256, 128, 1);
  gemm_bias_relu<<<dim3(8, P_TOT / 32), 256, 0, stream>>>(sp2, Wsp3, bsp3, sp3, P_TOT, REPRD, 256, 1);

  // pass-1 attention head -> w
  attn_fused<<<P_TOT / 32, 256, 0, stream>>>(0, node, node, 2 * ENCD, sp3,
      A1W, A1b, A2W, A2b, A3W, b3A, wbuf, REPRD, 0, 1, 0);
  adj_kernel<<<P_TOT, 256, 0, stream>>>(wbuf, Wadj, badj, adj);
  softmax_rows<<<NHEADS, 256, 0, stream>>>(adj, softA, NBOX, NBOX, 1);
  softmax_rows<<<NBOX, 256, 0, stream>>>(adj, softB, NHEADS, 1, NBOX);

  // message to humans (O head), reduce + LN -> hnodeL
  attn_fused<<<P_TOT / 32, 256, 0, stream>>>(1, node, nullptr, ENCD, sp3,
      O1W, O1b, O2W, O2b, O3W, b3O, wbuf, REPRD, 0, 0, 0);
  msg_reduce_ln<<<NHEADS, 256, 0, stream>>>(softA, wbuf, node, gln_h, bln_h, hnodeL, NBOX, 0);

  // message to objects (S head), reduce + LN -> nodeL
  attn_fused<<<P_TOT / 32, 256, 0, stream>>>(2, hnodeL, nullptr, ENCD, sp3,
      S1W, S1b, S2W, S2b, S3W, b3S, msgS, REPRD, 0, 0, 0);
  msg_reduce_ln<<<NBOX, 256, 0, stream>>>(softB, msgS, node, gln_o, bln_o, nodeL, NHEADS, 1);

  // final pair + global heads, compacted into concat output (18384 x 2048)
  attn_fused<<<P_TOT / 32, 256, 0, stream>>>(0, hnodeL, nodeL, 2 * ENCD, sp3,
      A1W, A1b, A2W, A2b, A3W, b3A, out, 2 * REPRD, 0, 1, 1);
  attn_fused<<<P_TOT / 32, 256, 0, stream>>>(3, gfeat, nullptr, 256, sp3,
      G1W, G1b, G2W, G2b, G3W, b3G, out, 2 * REPRD, REPRD, 1, 1);
}
